// AttentionLayer_26697516712216
// MI455X (gfx1250) — compile-verified, matrix operand order checked by reading, not run on hardware
//
#include <hip/hip_runtime.h>

// ---------------------------------------------------------------------------
// MI455X (gfx1250) fused multi-head attention, wave32 + WMMA f16->f32.
// B=8 T=12 S=512 D=128 H=8 hd=16.  All GEMMs via v_wmma_f32_16x16x32_f16.
// Round 2: attention processes 32 keys/iter so the P*V WMMA runs at full
// K=32 (no padding waste) with all 32 lanes loading real fragments.
// ---------------------------------------------------------------------------

typedef __attribute__((ext_vector_type(16))) _Float16 v16h;
typedef __attribute__((ext_vector_type(8)))  float    v8f;

#define DMODEL 128
#define SEQ    512
#define NH     8
#define HD     16
#define BT     96                       // B*T
#define NROWS  (BT * SEQ)               // 49152 token rows

__device__ __forceinline__ v8f zero_v8f() {
  v8f r;
#pragma unroll
  for (int i = 0; i < 8; ++i) r[i] = 0.0f;
  return r;
}
__device__ __forceinline__ v16h zero_v16h() {
  v16h r;
#pragma unroll
  for (int i = 0; i < 16; ++i) r[i] = (_Float16)0.0f;
  return r;
}

// ---------------------------------------------------------------------------
// Kernel 0: convert + transpose the 4 weight matrices to f16.
// Wt[n][k] = W[k][n] so WMMA B-fragments (lane = N column, contiguous K
// halves) are contiguous 32-byte loads.
// ---------------------------------------------------------------------------
__global__ __launch_bounds__(256) void prep_weights_kernel(
    const float* __restrict__ wq, const float* __restrict__ wk,
    const float* __restrict__ wv, const float* __restrict__ wo,
    _Float16* __restrict__ wt) {
  const float* w = (blockIdx.x == 0) ? wq
                 : (blockIdx.x == 1) ? wk
                 : (blockIdx.x == 2) ? wv : wo;
  _Float16* dst = wt + (size_t)blockIdx.x * DMODEL * DMODEL;
  for (int i = threadIdx.x; i < DMODEL * DMODEL; i += blockDim.x) {
    int k = i >> 7, n = i & 127;
    dst[(size_t)n * DMODEL + k] = (_Float16)w[i];
  }
}

// ---------------------------------------------------------------------------
// Kernel 1: fused Q/K/V projection GEMM: [49152 x 128] @ [128 x 128] + bias.
// grid = (384, 3); 8 waves/block, each wave owns a 16-row x 128-col strip,
// K = 128 in 4 steps of 32 (one wmma_f32_16x16x32_f16 per N-tile per step).
// Outputs (f16):
//   Q: row-major, pre-scaled by 1/sqrt(hd)=0.25 (A-fragment of Q*K^T)
//   K: row-major                                 (B-fragment of Q*K^T)
//   V: transposed per head Vt[bt][h][n][s]       (B-fragment of P*V)
// ---------------------------------------------------------------------------
__global__ __launch_bounds__(256) void qkv_proj_kernel(
    const float* __restrict__ xq, const float* __restrict__ xk,
    const float* __restrict__ xv, const _Float16* __restrict__ wt4,
    const float* __restrict__ bq, const float* __restrict__ bk,
    const float* __restrict__ bv, _Float16* __restrict__ qs,
    _Float16* __restrict__ ks, _Float16* __restrict__ vt) {
  const int proj = blockIdx.y;
  const float* x;
  const _Float16* wt;
  const float* bias;
  if (proj == 0)      { x = xq; wt = wt4;                     bias = bq; }
  else if (proj == 1) { x = xk; wt = wt4 + DMODEL * DMODEL;   bias = bk; }
  else                { x = xv; wt = wt4 + 2 * DMODEL*DMODEL; bias = bv; }

  const int wave = threadIdx.x >> 5;
  const int lane = threadIdx.x & 31;
  const int m    = lane & 15;          // A row / B column / D column
  const int hi   = lane >> 4;          // half-wave: K offset 0 or 16
  const int kofs = hi * 16;
  const int row0 = blockIdx.x * 128 + wave * 16;

  v8f c[8];
#pragma unroll
  for (int i = 0; i < 8; ++i) c[i] = zero_v8f();

  for (int k0 = 0; k0 < DMODEL; k0 += 32) {
    // A fragment: 16 f32 -> f16 converts, contiguous row chunk.
    const float* ap = x + (size_t)(row0 + m) * DMODEL + k0 + kofs;
    v16h a;
#pragma unroll
    for (int j = 0; j < 16; j += 4) {
      float4 f = *(const float4*)(ap + j);
      a[j + 0] = (_Float16)f.x; a[j + 1] = (_Float16)f.y;
      a[j + 2] = (_Float16)f.z; a[j + 3] = (_Float16)f.w;
    }
#pragma unroll
    for (int nt = 0; nt < 8; ++nt) {
      v16h b = *(const v16h*)(wt + (size_t)(nt * 16 + m) * DMODEL + k0 + kofs);
      c[nt] = __builtin_amdgcn_wmma_f32_16x16x32_f16(
          false, a, false, b, (short)0, c[nt], false, false);
    }
  }

#pragma unroll
  for (int nt = 0; nt < 8; ++nt) {
#pragma unroll
    for (int r = 0; r < 8; ++r) {
      const int row = row0 + r + hi * 8;          // C/D: reg r -> row r / r+8
      const int col = nt * 16 + m;                // lane -> column
      float v = c[nt][r] + bias[col];
      if (proj == 0) {
        qs[(size_t)row * DMODEL + col] = (_Float16)(v * 0.25f);
      } else if (proj == 1) {
        ks[(size_t)row * DMODEL + col] = (_Float16)v;
      } else {
        const int bt = row >> 9, s = row & 511;   // SEQ = 512
        const int h = nt, nn = m;                 // head-aligned 16-col tiles
        vt[((size_t)(bt * NH + h) * HD + nn) * SEQ + s] = (_Float16)v;
      }
    }
  }
}

// ---------------------------------------------------------------------------
// Kernel 2: flash-style attention per (bt, head).  grid = (96, 8),
// 8 waves/block, each wave processes 4 query tiles of 16 rows.
// Per 32-key block:
//   2x  S = Q*K^T WMMA (hd=16 zero-padded to K=32 -- unavoidable),
//   fused online softmax over 32 columns (shfl_xor within 16-lane halves,
//   matching the C/D layout: reg r <-> row r / r+8, lane <-> column),
//   P staged through a per-wave 16x32 LDS tile,
//   1x  O += P*V WMMA at FULL K=32: lanes 0-15 carry keys 0-15, lanes
//   16-31 carry keys 16-31 -- every lane loads a real 32-byte fragment.
// ---------------------------------------------------------------------------
__global__ __launch_bounds__(256) void attn_kernel(
    const _Float16* __restrict__ qs, const _Float16* __restrict__ ks,
    const _Float16* __restrict__ vt, _Float16* __restrict__ ctx) {
  __shared__ __align__(32) _Float16 scr[8][16][32];  // per-wave P bounce (8KB)

  const int bt   = blockIdx.x;
  const int h    = blockIdx.y;
  const int wave = threadIdx.x >> 5;
  const int lane = threadIdx.x & 31;
  const int m    = lane & 15;
  const int hi   = lane >> 4;

  const _Float16* qbase = qs + (size_t)bt * SEQ * DMODEL + h * HD;
  const _Float16* kbase = ks + (size_t)bt * SEQ * DMODEL + h * HD;
  const _Float16* vbase = vt + (size_t)(bt * NH + h) * HD * SEQ;
  // A-fragment read pointer: lane m -> row m, halves hi*16..hi*16+15.
  const _Float16* scrow = &scr[wave][m][hi * 16];

  for (int mt = wave; mt < SEQ / 16; mt += 8) {
    const int m0 = mt * 16;

    v16h qa = zero_v16h();                 // lanes 16..31 stay zero: K pad
    if (hi == 0) qa = *(const v16h*)(qbase + (size_t)(m0 + m) * DMODEL);

    v8f o = zero_v8f();
    float mrow[8], lrow[8];
#pragma unroll
    for (int r = 0; r < 8; ++r) { mrow[r] = -3.0e38f; lrow[r] = 0.0f; }

    for (int kb = 0; kb < SEQ / 32; ++kb) {      // 16 iterations of 32 keys
      v16h kf0 = zero_v16h();
      v16h kf1 = zero_v16h();
      if (hi == 0) {
        kf0 = *(const v16h*)(kbase + (size_t)(kb * 32 + m) * DMODEL);
        kf1 = *(const v16h*)(kbase + (size_t)(kb * 32 + 16 + m) * DMODEL);
      }
      v8f s0 = __builtin_amdgcn_wmma_f32_16x16x32_f16(
          false, qa, false, kf0, (short)0, zero_v8f(), false, false);
      v8f s1 = __builtin_amdgcn_wmma_f32_16x16x32_f16(
          false, qa, false, kf1, (short)0, zero_v8f(), false, false);

      // Online softmax over both tiles: reg r holds row (r + hi*8).
#pragma unroll
      for (int r = 0; r < 8; ++r) {
        const float v0 = s0[r];
        const float v1 = s1[r];
        float mx = fmaxf(v0, v1);
        mx = fmaxf(mx, __shfl_xor(mx, 1, 32));
        mx = fmaxf(mx, __shfl_xor(mx, 2, 32));
        mx = fmaxf(mx, __shfl_xor(mx, 4, 32));
        mx = fmaxf(mx, __shfl_xor(mx, 8, 32));
        const float mnew  = fmaxf(mrow[r], mx);
        const float scale = __expf(mrow[r] - mnew);
        const float p0    = __expf(v0 - mnew);
        const float p1    = __expf(v1 - mnew);
        float sum = p0 + p1;
        sum += __shfl_xor(sum, 1, 32);
        sum += __shfl_xor(sum, 2, 32);
        sum += __shfl_xor(sum, 4, 32);
        sum += __shfl_xor(sum, 8, 32);
        lrow[r] = lrow[r] * scale + sum;
        mrow[r] = mnew;
        o[r] *= scale;
        scr[wave][r + hi * 8][m]      = (_Float16)p0;   // D-layout -> LDS
        scr[wave][r + hi * 8][16 + m] = (_Float16)p1;
      }
      // Intra-wave LDS RAW: wait for this wave's DS stores, then read the
      // transposed 16x32 P tile as a full-K A-fragment.
      asm volatile("s_wait_dscnt 0" ::: "memory");

      v16h pa = *(const v16h*)scrow;                           // all 32 lanes
      v16h vf = *(const v16h*)(vbase + (size_t)m * SEQ + kb * 32 + hi * 16);
      o = __builtin_amdgcn_wmma_f32_16x16x32_f16(
          false, pa, false, vf, (short)0, o, false, false);
    }

#pragma unroll
    for (int r = 0; r < 8; ++r) {
      const float inv = 1.0f / lrow[r];
      const int row = m0 + r + hi * 8;
      ctx[(size_t)(bt * SEQ + row) * DMODEL + h * HD + m] =
          (_Float16)(o[r] * inv);
    }
  }
}

// ---------------------------------------------------------------------------
// Kernel 3: output projection GEMM: ctx(f16) @ Wo^T(f16) + bo -> f32 out.
// ---------------------------------------------------------------------------
__global__ __launch_bounds__(256) void out_proj_kernel(
    const _Float16* __restrict__ ctx, const _Float16* __restrict__ wot,
    const float* __restrict__ bo, float* __restrict__ out) {
  const int wave = threadIdx.x >> 5;
  const int lane = threadIdx.x & 31;
  const int m    = lane & 15;
  const int hi   = lane >> 4;
  const int kofs = hi * 16;
  const int row0 = blockIdx.x * 128 + wave * 16;

  v8f c[8];
#pragma unroll
  for (int i = 0; i < 8; ++i) c[i] = zero_v8f();

  for (int k0 = 0; k0 < DMODEL; k0 += 32) {
    v16h a = *(const v16h*)(ctx + (size_t)(row0 + m) * DMODEL + k0 + kofs);
#pragma unroll
    for (int nt = 0; nt < 8; ++nt) {
      v16h b = *(const v16h*)(wot + (size_t)(nt * 16 + m) * DMODEL + k0 + kofs);
      c[nt] = __builtin_amdgcn_wmma_f32_16x16x32_f16(
          false, a, false, b, (short)0, c[nt], false, false);
    }
  }
#pragma unroll
  for (int nt = 0; nt < 8; ++nt) {
#pragma unroll
    for (int r = 0; r < 8; ++r) {
      const int row = row0 + r + hi * 8;
      const int col = nt * 16 + m;
      out[(size_t)row * DMODEL + col] = c[nt][r] + bo[col];
    }
  }
}

// ---------------------------------------------------------------------------
// Host launcher.  Workspace layout (f16):
//   [0]        Wq^T,Wk^T,Wv^T,Wo^T   4*128*128      (131072 B)
//   [+131072]  Q  scaled             49152*128      (12.6 MB)
//   [+...]     K, Vt, ctx            (12.6 MB each) -> total ~48.1 MB
// ---------------------------------------------------------------------------
extern "C" void kernel_launch(void* const* d_in, const int* in_sizes, int n_in,
                              void* d_out, int out_size, void* d_ws,
                              size_t ws_size, hipStream_t stream) {
  (void)in_sizes; (void)n_in; (void)out_size; (void)ws_size;

  const float* xq = (const float*)d_in[0];
  const float* xk = (const float*)d_in[1];
  const float* xv = (const float*)d_in[2];
  const float* Wq = (const float*)d_in[3];
  const float* bq = (const float*)d_in[4];
  const float* Wk = (const float*)d_in[5];
  const float* bk = (const float*)d_in[6];
  const float* Wv = (const float*)d_in[7];
  const float* bv = (const float*)d_in[8];
  const float* Wo = (const float*)d_in[9];
  const float* bo = (const float*)d_in[10];
  float* out = (float*)d_out;

  char* ws = (char*)d_ws;
  const size_t WT_BYTES  = 4 * (size_t)DMODEL * DMODEL * sizeof(_Float16);
  const size_t MAT_BYTES = (size_t)NROWS * DMODEL * sizeof(_Float16);

  _Float16* wt4 = (_Float16*)(ws);
  _Float16* qsb = (_Float16*)(ws + WT_BYTES);
  _Float16* ksb = (_Float16*)(ws + WT_BYTES + MAT_BYTES);
  _Float16* vtb = (_Float16*)(ws + WT_BYTES + 2 * MAT_BYTES);
  _Float16* ctx = (_Float16*)(ws + WT_BYTES + 3 * MAT_BYTES);
  _Float16* wot = wt4 + 3 * DMODEL * DMODEL;

  prep_weights_kernel<<<4, 256, 0, stream>>>(Wq, Wk, Wv, Wo, wt4);

  qkv_proj_kernel<<<dim3(NROWS / 128, 3), 256, 0, stream>>>(
      xq, xk, xv, wt4, bq, bk, bv, qsb, ksb, vtb);

  attn_kernel<<<dim3(BT, NH), 256, 0, stream>>>(qsb, ksb, vtb, ctx);

  out_proj_kernel<<<NROWS / 128, 256, 0, stream>>>(ctx, wot, bo, out);
}
